// DecoderLayer_73478300500666
// MI455X (gfx1250) — matrix operand order repaired and checked by reading, not compiled, not run on hardware
//
#include <hip/hip_runtime.h>
#include <hip/hip_bf16.h>

// ---------------------------------------------------------------------------
// Transformer decoder layer for MI455X (gfx1250, wave32).
//  - All matrix math: v_wmma_f32_16x16x32_bf16 (fp32 accumulate)
//  - All matrix tile staging: Tensor Data Mover (tensor_load_to_lds) with
//    bf16 operands pre-materialized in global memory (halves HBM traffic).
// ---------------------------------------------------------------------------

#define DMODEL 1024
#define NHEAD  16
#define HD     64
#define LSEQ   512
#define NBATCH 16
#define MROWS  (NBATCH * LSEQ)   // 8192

typedef __bf16 bf16_t;
typedef __attribute__((ext_vector_type(16))) __bf16 v16bf;
typedef __attribute__((ext_vector_type(8)))  __bf16 v8bf;
typedef __attribute__((ext_vector_type(8)))  float  v8f;
typedef __attribute__((ext_vector_type(4))) unsigned int u32x4;
typedef __attribute__((ext_vector_type(8))) int          i32x8;
typedef __attribute__((ext_vector_type(4))) int          i32x4;

__device__ __forceinline__ v8f wmma_bf16(v16bf a, v16bf b, v8f c) {
  return __builtin_amdgcn_wmma_f32_16x16x32_bf16(false, a, false, b, (short)0, c,
                                                 false, false);
}

// ---------------------------------------------------------------------------
// TDM: load a 2-D tile (tile_k bf16 along dim0, tile_rows rows, row stride
// row_stride elements) from global bf16 memory into LDS at lds_off.
// Descriptor per CDNA5 ISA ch.8: group0 {count,lds,global,type}, group1
// {data_size=2B, tensor dims, tile dims, dim0 stride}.
// ---------------------------------------------------------------------------
__device__ __forceinline__ void tdm_load_2d(unsigned lds_off, const bf16_t* gptr,
                                            unsigned tile_k, unsigned tile_rows,
                                            unsigned row_stride,
                                            unsigned tensor_d0, unsigned tensor_d1) {
  unsigned long long ga = (unsigned long long)gptr;
  u32x4 g0;
  g0[0] = 1u;                                   // count=1, user-mode, no gather
  g0[1] = lds_off;                              // lds_addr (bytes)
  g0[2] = (unsigned)(ga & 0xffffffffu);         // global_addr[31:0]
  g0[3] = (unsigned)((ga >> 32) & 0x1ffffffu) | (2u << 30);  // addr[56:32]|type=2
  i32x8 g1;
  g1[0] = (int)(1u << 16);                      // data_size=1 -> 2 bytes
  g1[1] = (int)((tensor_d0 & 0xffffu) << 16);   // tensor_dim0[15:0]
  g1[2] = (int)((tensor_d0 >> 16) | ((tensor_d1 & 0xffffu) << 16));
  g1[3] = (int)((tensor_d1 >> 16) | ((tile_k & 0xffffu) << 16)); // tile_dim0
  g1[4] = (int)(tile_rows & 0xffffu);           // tile_dim1 (tile_dim2=0)
  g1[5] = (int)row_stride;                      // tensor_dim0_stride[31:0]
  g1[6] = 0;
  g1[7] = 0;
  i32x4 z4 = {0, 0, 0, 0};
#if __clang_major__ >= 23
  i32x8 z8 = {0, 0, 0, 0, 0, 0, 0, 0};
  __builtin_amdgcn_tensor_load_to_lds(g0, g1, z4, z4, z8, 0);
#else
  __builtin_amdgcn_tensor_load_to_lds(g0, g1, z4, z4, 0);
#endif
}

__device__ __forceinline__ unsigned lds_offset_of(const void* p) {
  return (unsigned)(unsigned long long)(size_t)p;  // low 32 bits = LDS offset
}

// A-fragment (16x32 bf16): lane(l&15)=row M; lanes0-15 K 0..7/16..23,
// lanes16-31 K 8..15/24..31 (ISA 7.12.2).
__device__ __forceinline__ v16bf load_frag_a(const bf16_t* tile, int row0,
                                             int ldk, int kofs) {
  const int lane = threadIdx.x & 31;
  const bf16_t* p = tile + (size_t)(row0 + (lane & 15)) * ldk + kofs +
                    ((lane & 16) ? 8 : 0);
  v8bf lo = *(const v8bf*)p;
  v8bf hi = *(const v8bf*)(p + 16);
  v16bf r;
#pragma unroll
  for (int i = 0; i < 8; ++i) { r[i] = lo[i]; r[i + 8] = hi[i]; }
  return r;
}

// B-fragment (32x16 bf16) from transposed tile BT[n][k]: lane(l&15)=col N;
// lanes0-15 K 0..7/16..23, lanes16-31 K 8..15/24..31.
__device__ __forceinline__ v16bf load_frag_b(const bf16_t* tileT, int col0,
                                             int ldk, int kofs) {
  const int lane = threadIdx.x & 31;
  const bf16_t* p = tileT + (size_t)(col0 + (lane & 15)) * ldk + kofs +
                    ((lane & 16) ? 8 : 0);
  v8bf lo = *(const v8bf*)p;
  v8bf hi = *(const v8bf*)(p + 16);
  v16bf r;
#pragma unroll
  for (int i = 0; i < 8; ++i) { r[i] = lo[i]; r[i + 8] = hi[i]; }
  return r;
}

// ---------------------------------------------------------------------------
// One-time data layout kernels (bf16 mirrors).
// ---------------------------------------------------------------------------
__global__ __launch_bounds__(256)
void cvt_bf16_kernel(const float* __restrict__ in, bf16_t* __restrict__ out,
                     size_t n) {
  size_t i = (size_t)blockIdx.x * 1024 + threadIdx.x;
#pragma unroll
  for (int j = 0; j < 4; ++j, i += 256)
    if (i < n) out[i] = (bf16_t)in[i];
}

// WT[n][k] = (bf16)W[k][n]   (K=N=1024)
__global__ __launch_bounds__(256)
void wtrans_kernel(const float* __restrict__ W, bf16_t* __restrict__ WT) {
  int idx = blockIdx.x * 256 + threadIdx.x;
  int n = idx >> 10, k = idx & 1023;
  WT[idx] = (bf16_t)W[(size_t)k * DMODEL + n];
}

// vt[b][h][d][key] = (bf16)v[b][key][h*64+d]
__global__ __launch_bounds__(256)
void vtrans_kernel(const float* __restrict__ v, bf16_t* __restrict__ vt) {
  size_t idx = (size_t)blockIdx.x * 256 + threadIdx.x;
  int key = (int)(idx & (LSEQ - 1));
  int d   = (int)((idx >> 9) & (HD - 1));
  int h   = (int)((idx >> 15) & (NHEAD - 1));
  int b   = (int)(idx >> 19);
  vt[idx] = (bf16_t)v[((size_t)b * LSEQ + key) * DMODEL + h * HD + d];
}

// ---------------------------------------------------------------------------
// C[M,N] = act(A[M,K](bf16) @ WT[N,K]^T(bf16) + bias); optional bf16 copy
// Cb = (bf16)(C * cbScale).  TDM-staged 128x128 block tile, K depth 64.
// ---------------------------------------------------------------------------
__global__ __launch_bounds__(256)
void gemm_bias_kernel(const bf16_t* __restrict__ A, const bf16_t* __restrict__ WT,
                      const float* __restrict__ bias, float* __restrict__ C,
                      bf16_t* __restrict__ Cb, float cbScale,
                      int M, int K, int N, int act) {
  __shared__ alignas(16) bf16_t sA[128][64];   // 16 KB
  __shared__ alignas(16) bf16_t sB[128][64];   // 16 KB  (WT rows = output cols)
  const int bm = blockIdx.x * 128, bn = blockIdx.y * 128;
  const int tid = threadIdx.x, lane = tid & 31, wid = tid >> 5;
  const int wm = wid & 3, wn = wid >> 2;
  const int c16 = lane & 15, r8 = (lane >> 4) << 3;
  const unsigned ldsA = lds_offset_of(&sA[0][0]);
  const unsigned ldsB = lds_offset_of(&sB[0][0]);
  v8f acc[2][4] = {};

  for (int kk = 0; kk < K; kk += 64) {
    if (wid == 0) {
      tdm_load_2d(ldsA, A + (size_t)bm * K + kk, 64, 128, K, K, M);
      tdm_load_2d(ldsB, WT + (size_t)bn * K + kk, 64, 128, K, K, N);
      __builtin_amdgcn_s_wait_tensorcnt(0);
    }
    __syncthreads();
#pragma unroll
    for (int k2 = 0; k2 < 64; k2 += 32) {
      v16bf bfrag[4];
#pragma unroll
      for (int ni = 0; ni < 4; ++ni)
        bfrag[ni] = load_frag_b(&sB[0][0], wn * 64 + ni * 16, 64, k2);
#pragma unroll
      for (int mi = 0; mi < 2; ++mi) {
        v16bf afrag = load_frag_a(&sA[0][0], wm * 32 + mi * 16, 64, k2);
#pragma unroll
        for (int ni = 0; ni < 4; ++ni)
          acc[mi][ni] = wmma_bf16(afrag, bfrag[ni], acc[mi][ni]);
      }
    }
    __syncthreads();
  }

#pragma unroll
  for (int mi = 0; mi < 2; ++mi)
#pragma unroll
    for (int ni = 0; ni < 4; ++ni) {
      int col = bn + wn * 64 + ni * 16 + c16;
      float bv = bias[col];
#pragma unroll
      for (int r = 0; r < 8; ++r) {
        int row = bm + wm * 32 + mi * 16 + r8 + r;
        float x = acc[mi][ni][r] + bv;
        if (act) x = 0.5f * x * (1.0f + erff(x * 0.70710678118f));
        C[(size_t)row * N + col] = x;
        if (Cb) Cb[(size_t)row * N + col] = (bf16_t)(x * cbScale);
      }
    }
}

// ---------------------------------------------------------------------------
// attention_vis: out[b,i,j] = scale * sum_k A[b,i,k]*B[b,j,k]  (bf16 NT GEMM)
// ---------------------------------------------------------------------------
__global__ __launch_bounds__(256)
void vis_gemm_nt_kernel(const bf16_t* __restrict__ Ab, const bf16_t* __restrict__ Bb,
                        float* __restrict__ out, int L, int K, float scale) {
  __shared__ alignas(16) bf16_t sA[128][64];
  __shared__ alignas(16) bf16_t sB[128][64];
  const int b = blockIdx.z;
  const int bm = blockIdx.x * 128, bn = blockIdx.y * 128;
  const int tid = threadIdx.x, lane = tid & 31, wid = tid >> 5;
  const int wm = wid & 3, wn = wid >> 2;
  const int c16 = lane & 15, r8 = (lane >> 4) << 3;
  const unsigned ldsA = lds_offset_of(&sA[0][0]);
  const unsigned ldsB = lds_offset_of(&sB[0][0]);
  const bf16_t* Abase = Ab + ((size_t)b * L + bm) * K;
  const bf16_t* Bbase = Bb + ((size_t)b * L + bn) * K;
  v8f acc[2][4] = {};

  for (int kk = 0; kk < K; kk += 64) {
    if (wid == 0) {
      tdm_load_2d(ldsA, Abase + kk, 64, 128, K, K, L);
      tdm_load_2d(ldsB, Bbase + kk, 64, 128, K, K, L);
      __builtin_amdgcn_s_wait_tensorcnt(0);
    }
    __syncthreads();
#pragma unroll
    for (int k2 = 0; k2 < 64; k2 += 32) {
      v16bf bfrag[4];
#pragma unroll
      for (int ni = 0; ni < 4; ++ni)
        bfrag[ni] = load_frag_b(&sB[0][0], wn * 64 + ni * 16, 64, k2);
#pragma unroll
      for (int mi = 0; mi < 2; ++mi) {
        v16bf afrag = load_frag_a(&sA[0][0], wm * 32 + mi * 16, 64, k2);
#pragma unroll
        for (int ni = 0; ni < 4; ++ni)
          acc[mi][ni] = wmma_bf16(afrag, bfrag[ni], acc[mi][ni]);
      }
    }
    __syncthreads();
  }

#pragma unroll
  for (int mi = 0; mi < 2; ++mi)
#pragma unroll
    for (int ni = 0; ni < 4; ++ni) {
      int col = bn + wn * 64 + ni * 16 + c16;
#pragma unroll
      for (int r = 0; r < 8; ++r) {
        int row = bm + wm * 32 + mi * 16 + r8 + r;
        out[((size_t)b * L + row) * L + col] = acc[mi][ni][r] * scale;
      }
    }
}

// ---------------------------------------------------------------------------
// Flash attention. qs: bf16, pre-scaled by 1/sqrt(64); ks: bf16 (b,l,D);
// vt: bf16 (b,h,d,key). One block per (128 q-rows, head, batch); each of the
// 8 waves owns 16 full rows (in-wave softmax). K/V tiles TDM-staged.
// ---------------------------------------------------------------------------
__global__ __launch_bounds__(256)
void attn_kernel(const bf16_t* __restrict__ qs, const bf16_t* __restrict__ ks,
                 const bf16_t* __restrict__ vt, const float* __restrict__ dec_mask,
                 float* __restrict__ ctx, int Lq, int Lk, int causal) {
  __shared__ alignas(16) bf16_t sQ[128][HD];   // 16 KB
  __shared__ alignas(16) bf16_t sK[HD][HD];    //  8 KB  [key][d]
  __shared__ alignas(16) bf16_t sV[HD][HD];    //  8 KB  [d][key]
  __shared__ alignas(16) bf16_t sP[128][HD];   // 16 KB  (wave-private rows)
  const int b = blockIdx.z, h = blockIdx.y, qb = blockIdx.x * 128;
  const int tid = threadIdx.x, lane = tid & 31, wid = tid >> 5;
  const int c16 = lane & 15, r8 = (lane >> 4) << 3;
  const unsigned ldsQ = lds_offset_of(&sQ[0][0]);
  const unsigned ldsK = lds_offset_of(&sK[0][0]);
  const unsigned ldsV = lds_offset_of(&sV[0][0]);

  const bf16_t* qbase = qs + ((size_t)b * Lq + qb) * DMODEL + h * HD;
  const bf16_t* kbase = ks + (size_t)b * Lk * DMODEL + h * HD;
  const bf16_t* vbase = vt + (size_t)(b * NHEAD + h) * HD * Lk;

  if (wid == 0)
    tdm_load_2d(ldsQ, qbase, HD, 128, DMODEL, DMODEL, Lq);

  float m[8], l[8];
  v8f o[4] = {};
#pragma unroll
  for (int r = 0; r < 8; ++r) { m[r] = -1e30f; l[r] = 0.0f; }

  for (int kb = 0; kb < Lk; kb += 64) {
    if (wid == 0) {
      tdm_load_2d(ldsK, kbase + (size_t)kb * DMODEL, HD, HD, DMODEL, DMODEL, Lk);
      tdm_load_2d(ldsV, vbase + kb, HD, HD, Lk, Lk, HD);
      __builtin_amdgcn_s_wait_tensorcnt(0);    // covers Q on first iteration
    }
    __syncthreads();

    // S(16x64 per wave) = Qs @ K^T
    v8f s[4] = {};
#pragma unroll
    for (int k2 = 0; k2 < 64; k2 += 32) {
      v16bf aq = load_frag_a(&sQ[0][0], wid * 16, HD, k2);
#pragma unroll
      for (int ni = 0; ni < 4; ++ni) {
        v16bf bk = load_frag_b(&sK[0][0], ni * 16, HD, k2);
        s[ni] = wmma_bf16(aq, bk, s[ni]);
      }
    }

    if (causal) {
#pragma unroll
      for (int ni = 0; ni < 4; ++ni) {
        int kc = kb + ni * 16 + c16;
        float dm = dec_mask[(size_t)b * Lk + kc];
#pragma unroll
        for (int r = 0; r < 8; ++r) {
          int qr = qb + wid * 16 + r8 + r;
          float mk = (kc <= qr) ? dm : 0.0f;
          s[ni][r] += (1.0f - mk) * -10000.0f;
        }
      }
    }

    // online softmax (rows: 8 per lane; cols: across 16-lane half-groups)
    float rmax[8];
#pragma unroll
    for (int r = 0; r < 8; ++r)
      rmax[r] = fmaxf(fmaxf(s[0][r], s[1][r]), fmaxf(s[2][r], s[3][r]));
#pragma unroll
    for (int off = 1; off < 16; off <<= 1)
#pragma unroll
      for (int r = 0; r < 8; ++r)
        rmax[r] = fmaxf(rmax[r], __shfl_xor(rmax[r], off, 32));

    float corr[8], rs[8];
#pragma unroll
    for (int r = 0; r < 8; ++r) {
      float mn = fmaxf(m[r], rmax[r]);
      corr[r] = __expf(m[r] - mn);
      m[r] = mn;
      rs[r] = 0.0f;
    }
#pragma unroll
    for (int ni = 0; ni < 4; ++ni)
#pragma unroll
      for (int r = 0; r < 8; ++r) {
        float p = __expf(s[ni][r] - m[r]);
        rs[r] += p;
        sP[wid * 16 + r8 + r][ni * 16 + c16] = (bf16_t)p;
      }
#pragma unroll
    for (int off = 1; off < 16; off <<= 1)
#pragma unroll
      for (int r = 0; r < 8; ++r) rs[r] += __shfl_xor(rs[r], off, 32);
#pragma unroll
    for (int r = 0; r < 8; ++r) l[r] = l[r] * corr[r] + rs[r];
#pragma unroll
    for (int j = 0; j < 4; ++j)
#pragma unroll
      for (int r = 0; r < 8; ++r) o[j][r] *= corr[r];

    // O += P @ V  (sP rows are wave-private; same-wave LDS RAW is in-order)
#pragma unroll
    for (int k2 = 0; k2 < 64; k2 += 32) {
      v16bf ap = load_frag_a(&sP[0][0], wid * 16, HD, k2);
#pragma unroll
      for (int j = 0; j < 4; ++j) {
        v16bf bv = load_frag_b(&sV[0][0], j * 16, HD, k2);
        o[j] = wmma_bf16(ap, bv, o[j]);
      }
    }
    __syncthreads();   // all reads of sK/sV done before next TDM overwrite
  }

  float* cbase = ctx + ((size_t)b * Lq + qb) * DMODEL + h * HD;
#pragma unroll
  for (int j = 0; j < 4; ++j)
#pragma unroll
    for (int r = 0; r < 8; ++r) {
      int qr = wid * 16 + r8 + r;
      cbase[(size_t)qr * DMODEL + j * 16 + c16] = o[j][r] / l[r];
    }
}

// ---------------------------------------------------------------------------
// out[row,:] = layernorm(a+b)*g+beta ; optional bf16 mirror for GEMM chaining
// ---------------------------------------------------------------------------
__global__ __launch_bounds__(256)
void addln_kernel(const float* __restrict__ a, const float* __restrict__ b,
                  const float* __restrict__ g, const float* __restrict__ beta,
                  float* __restrict__ out, bf16_t* __restrict__ outb) {
  const size_t row = blockIdx.x;
  const int tid = threadIdx.x;
  __shared__ float red[256];
  const float* pa = a + row * DMODEL;
  const float* pb = b + row * DMODEL;
  float vals[4];
  float s = 0.0f;
#pragma unroll
  for (int i = 0; i < 4; ++i) {
    vals[i] = pa[tid + i * 256] + pb[tid + i * 256];
    s += vals[i];
  }
  red[tid] = s; __syncthreads();
  for (int off = 128; off > 0; off >>= 1) {
    if (tid < off) red[tid] += red[tid + off];
    __syncthreads();
  }
  float mu = red[0] * (1.0f / DMODEL);
  __syncthreads();
  float vsum = 0.0f;
#pragma unroll
  for (int i = 0; i < 4; ++i) { float d = vals[i] - mu; vsum += d * d; }
  red[tid] = vsum; __syncthreads();
  for (int off = 128; off > 0; off >>= 1) {
    if (tid < off) red[tid] += red[tid + off];
    __syncthreads();
  }
  float inv = rsqrtf(red[0] * (1.0f / DMODEL) + 1e-6f);
#pragma unroll
  for (int i = 0; i < 4; ++i) {
    int c = tid + i * 256;
    float y = (vals[i] - mu) * inv * g[c] + beta[c];
    out[row * DMODEL + c] = y;
    if (outb) outb[row * DMODEL + c] = (bf16_t)y;
  }
}

// ---------------------------------------------------------------------------
extern "C" void kernel_launch(void* const* d_in, const int* in_sizes, int n_in,
                              void* d_out, int out_size, void* d_ws, size_t ws_size,
                              hipStream_t stream) {
  const float* x        = (const float*)d_in[0];
  const float* dec_mask = (const float*)d_in[1];
  const float* enc      = (const float*)d_in[2];
  const float* sa_wq = (const float*)d_in[3],  *sa_bq = (const float*)d_in[4];
  const float* sa_wk = (const float*)d_in[5],  *sa_bk = (const float*)d_in[6];
  const float* sa_wv = (const float*)d_in[7],  *sa_bv = (const float*)d_in[8];
  const float* n1_g  = (const float*)d_in[9],  *n1_b  = (const float*)d_in[10];
  const float* ca_wq = (const float*)d_in[11], *ca_bq = (const float*)d_in[12];
  const float* ca_wk = (const float*)d_in[13], *ca_bk = (const float*)d_in[14];
  const float* ca_wv = (const float*)d_in[15], *ca_bv = (const float*)d_in[16];
  const float* n2_g  = (const float*)d_in[17], *n2_b  = (const float*)d_in[18];
  const float* int_w = (const float*)d_in[19], *int_b = (const float*)d_in[20];
  const float* out_w = (const float*)d_in[21], *out_b = (const float*)d_in[22];
  const float* out_g = (const float*)d_in[23], *out_beta = (const float*)d_in[24];

  float* out = (float*)d_out;                          // (16,512,1024)
  float* vis = out + (size_t)NBATCH * LSEQ * DMODEL;   // (16,512,512)

  const size_t SLOT_F = (size_t)MROWS * DMODEL;        // fp32 activation slot
  const size_t SLOT_H = (size_t)MROWS * DMODEL;        // bf16 activation slot
  const size_t WSLOT  = (size_t)DMODEL * DMODEL;       // bf16 weight slot
  float* ws = (float*)d_ws;
  float* s0 = ws + 0 * SLOT_F;   // q / cq / inter
  float* s1 = ws + 1 * SLOT_F;   // k / ck / y
  float* s2 = ws + 2 * SLOT_F;   // v / cv
  float* s3 = ws + 3 * SLOT_F;   // sa_ctx / ca_ctx
  float* s4 = ws + 4 * SLOT_F;   // h
  float* s5 = ws + 5 * SLOT_F;   // h2
  bf16_t* hb16 = (bf16_t*)(ws + 6 * SLOT_F);
  bf16_t* xb    = hb16 + 0 * SLOT_H;   // bf16(x)
  bf16_t* encb  = hb16 + 1 * SLOT_H;   // bf16(enc)
  bf16_t* qkb   = hb16 + 2 * SLOT_H;   // bf16 q (pre-scaled) / cq
  bf16_t* kkb   = hb16 + 3 * SLOT_H;   // bf16 k / ck
  bf16_t* vtb   = hb16 + 4 * SLOT_H;   // bf16 V^T (b,h,d,key)
  bf16_t* hb    = hb16 + 5 * SLOT_H;   // bf16 h
  bf16_t* h2b   = hb16 + 6 * SLOT_H;   // bf16 h2
  bf16_t* intb  = hb16 + 7 * SLOT_H;   // bf16 inter
  bf16_t* wtb   = hb16 + 8 * SLOT_H;
  bf16_t* wt_saq = wtb + 0 * WSLOT, *wt_sak = wtb + 1 * WSLOT;
  bf16_t* wt_sav = wtb + 2 * WSLOT, *wt_caq = wtb + 3 * WSLOT;
  bf16_t* wt_cak = wtb + 4 * WSLOT, *wt_cav = wtb + 5 * WSLOT;
  bf16_t* wt_int = wtb + 6 * WSLOT, *wt_out = wtb + 7 * WSLOT;

  const dim3 blk(256);
  const dim3 gGemm(MROWS / 128, DMODEL / 128);
  const dim3 gAttn(LSEQ / 128, NHEAD, NBATCH);
  const dim3 gVis(LSEQ / 128, LSEQ / 128, NBATCH);
  const int gCvt = (int)(SLOT_F / 1024);
  const int gWT  = (int)(WSLOT / 256);
  const int gVT  = (int)(SLOT_F / 256);
  const float QSCALE = 0.125f;  // 1/sqrt(64)

  // ---- one-time bf16 layout ----
  cvt_bf16_kernel<<<gCvt, blk, 0, stream>>>(x,   xb,   SLOT_F);
  cvt_bf16_kernel<<<gCvt, blk, 0, stream>>>(enc, encb, SLOT_F);
  wtrans_kernel<<<gWT, blk, 0, stream>>>(sa_wq, wt_saq);
  wtrans_kernel<<<gWT, blk, 0, stream>>>(sa_wk, wt_sak);
  wtrans_kernel<<<gWT, blk, 0, stream>>>(sa_wv, wt_sav);
  wtrans_kernel<<<gWT, blk, 0, stream>>>(ca_wq, wt_caq);
  wtrans_kernel<<<gWT, blk, 0, stream>>>(ca_wk, wt_cak);
  wtrans_kernel<<<gWT, blk, 0, stream>>>(ca_wv, wt_cav);
  wtrans_kernel<<<gWT, blk, 0, stream>>>(int_w, wt_int);
  wtrans_kernel<<<gWT, blk, 0, stream>>>(out_w, wt_out);

  // ---- self attention ----
  gemm_bias_kernel<<<gGemm, blk, 0, stream>>>(xb, wt_saq, sa_bq, s0, qkb, QSCALE,
                                              MROWS, DMODEL, DMODEL, 0);
  gemm_bias_kernel<<<gGemm, blk, 0, stream>>>(xb, wt_sak, sa_bk, s1, kkb, 1.0f,
                                              MROWS, DMODEL, DMODEL, 0);
  gemm_bias_kernel<<<gGemm, blk, 0, stream>>>(xb, wt_sav, sa_bv, s2, nullptr, 1.0f,
                                              MROWS, DMODEL, DMODEL, 0);
  vtrans_kernel<<<gVT, blk, 0, stream>>>(s2, vtb);
  attn_kernel<<<gAttn, blk, 0, stream>>>(qkb, kkb, vtb, dec_mask, s3, LSEQ, LSEQ, 1);
  addln_kernel<<<MROWS, blk, 0, stream>>>(s3, x, n1_g, n1_b, s4, hb);

  // ---- cross attention ----
  gemm_bias_kernel<<<gGemm, blk, 0, stream>>>(hb, wt_caq, ca_bq, s0, qkb, QSCALE,
                                              MROWS, DMODEL, DMODEL, 0);
  gemm_bias_kernel<<<gGemm, blk, 0, stream>>>(encb, wt_cak, ca_bk, s1, kkb, 1.0f,
                                              MROWS, DMODEL, DMODEL, 0);
  gemm_bias_kernel<<<gGemm, blk, 0, stream>>>(encb, wt_cav, ca_bv, s2, nullptr, 1.0f,
                                              MROWS, DMODEL, DMODEL, 0);
  vtrans_kernel<<<gVT, blk, 0, stream>>>(s2, vtb);
  attn_kernel<<<gAttn, blk, 0, stream>>>(qkb, kkb, vtb, dec_mask, s3, LSEQ, LSEQ, 0);
  // vis = (cq·ck over D)/(H*sqrt(hd)); qkb is pre-scaled by 0.125 -> 1/16 left
  vis_gemm_nt_kernel<<<gVis, blk, 0, stream>>>(qkb, kkb, vis, LSEQ, DMODEL, 0.0625f);
  addln_kernel<<<MROWS, blk, 0, stream>>>(s4, s3, n2_g, n2_b, s5, h2b);

  // ---- FFN ----
  gemm_bias_kernel<<<gGemm, blk, 0, stream>>>(h2b, wt_int, int_b, s0, intb, 1.0f,
                                              MROWS, DMODEL, DMODEL, 1);
  gemm_bias_kernel<<<gGemm, blk, 0, stream>>>(intb, wt_out, out_b, s1, nullptr, 1.0f,
                                              MROWS, DMODEL, DMODEL, 0);
  addln_kernel<<<MROWS, blk, 0, stream>>>(s1, s0, out_g, out_beta, out, nullptr);
}
